// LogitsNTXentLoss_88167088652546
// MI455X (gfx1250) — matrix operand order repaired and checked by reading, not compiled, not run on hardware
//
#include <hip/hip_runtime.h>
#include <hip/hip_bf16.h>

typedef __attribute__((ext_vector_type(16))) _Float16 v16h;
typedef __attribute__((ext_vector_type(8)))  _Float16 v8h;
typedef __attribute__((ext_vector_type(4)))  _Float16 v4h;
typedef __attribute__((ext_vector_type(8)))  float    v8f;

#define NROWS_HALF 4096          // N
#define NROWS      8192          // 2N
#define DIM        1024          // D
#define INV_T      (1.0f / 0.07f)
#define MT         2             // row tiles (16 rows each) per block
#define NT         4             // column tiles per wave group
#define ROWS_PB    (MT * 16)     // 32 rows per block
#define NRB        (NROWS / ROWS_PB)   // 256 row-blocks
#define NCT        (NROWS / 16)        // 512 column tiles
#define NGRP       (NCT / NT)          // 128 column groups of 4 tiles

// ---------------------------------------------------------------------------
// Kernel 1: L2-normalize each row of [logits1; logits2] -> f16 z in workspace
// ---------------------------------------------------------------------------
__global__ void ntx_normalize(const float* __restrict__ x1,
                              const float* __restrict__ x2,
                              _Float16* __restrict__ z) {
  const int row  = blockIdx.x;
  const int tid  = threadIdx.x;
  const int lane = tid & 31;
  const int wave = tid >> 5;

  const float* src = (row < NROWS_HALF)
                       ? (x1 + (size_t)row * DIM)
                       : (x2 + (size_t)(row - NROWS_HALF) * DIM);

  float4 v = ((const float4*)src)[tid];           // 256 * 4 = 1024
  float ss = v.x * v.x + v.y * v.y + v.z * v.z + v.w * v.w;

  #pragma unroll
  for (int off = 16; off > 0; off >>= 1)
    ss += __shfl_xor(ss, off, 32);

  __shared__ float wsum[8];
  __shared__ float s_inv;
  if (lane == 0) wsum[wave] = ss;
  __syncthreads();
  if (tid == 0) {
    float t = 0.0f;
    #pragma unroll
    for (int w = 0; w < 8; ++w) t += wsum[w];
    s_inv = rsqrtf(t);
  }
  __syncthreads();
  const float inv = s_inv;

  v4h o;
  o[0] = (_Float16)(v.x * inv);
  o[1] = (_Float16)(v.y * inv);
  o[2] = (_Float16)(v.z * inv);
  o[3] = (_Float16)(v.w * inv);
  ((v4h*)(z + (size_t)row * DIM))[tid] = o;
}

// ---------------------------------------------------------------------------
// Kernel 2: fused Gram + masked logsumexp, register-tiled.
// Block = 8 wave32s, 32 rows (2 row-tiles of 16) staged in 64KB dynamic LDS.
// Each wave sweeps groups of 4 consecutive column tiles; per K-slice it loads
// 2 A-frags (LDS) + 4 B-frags (global, immediate-offset clauses) and issues
// 8 independent v_wmma_f32_16x16x32_f16 chains.
//   B bytes/WMMA: 0.5KB/wave, A LDS bytes/WMMA: 0.25KB/wave -> XDL stays fed.
// ---------------------------------------------------------------------------
__global__ void ntx_gram_lse(const _Float16* __restrict__ z,
                             float* __restrict__ blockLoss) {
  extern __shared__ __align__(16) _Float16 aTile[];   // ROWS_PB * DIM = 64 KB
  __shared__ float waveSums[8][ROWS_PB];
  __shared__ float posArr[ROWS_PB];
  __shared__ float lmp[ROWS_PB];

  const int tid  = threadIdx.x;
  const int lane = tid & 31;
  const int wave = tid >> 5;
  const int h    = lane >> 4;   // half-wave select
  const int r    = lane & 15;   // row (A) / col (B) within a 16-tile

  const int rb      = blockIdx.x;            // 0..NRB-1
  const int rowBase = rb * ROWS_PB;

  // Stage A tile: 32*1024 halves = 4096 x v8h chunks; 16 per thread.
  {
    const v8h* src = (const v8h*)(z + (size_t)rowBase * DIM);
    v8h* dst = (v8h*)aTile;
    #pragma unroll
    for (int i = 0; i < 16; ++i)
      dst[tid + i * 256] = src[tid + i * 256];
  }
  __syncthreads();

  float partial[MT][8];
  #pragma unroll
  for (int mt = 0; mt < MT; ++mt)
    #pragma unroll
    for (int v = 0; v < 8; ++v) partial[mt][v] = 0.0f;

  for (int g = wave; g < NGRP; g += 8) {
    // base pointer for this lane's B columns; tiles nt are +nt*16 rows ahead
    const _Float16* bcol = z + (size_t)(g * (NT * 16) + r) * DIM;

    if (g + 8 < NGRP)  // hint next group (global_prefetch_b8)
      __builtin_prefetch(z + (size_t)((g + 8) * (NT * 16) + r) * DIM, 0, 1);

    v8f c[MT][NT];
    #pragma unroll
    for (int mt = 0; mt < MT; ++mt)
      #pragma unroll
      for (int nt = 0; nt < NT; ++nt) c[mt][nt] = (v8f){};

    #pragma unroll 2
    for (int k0 = 0; k0 < DIM; k0 += 32) {
      union { v16h v; v8h p[2]; } a[MT], b[NT];
      #pragma unroll
      for (int mt = 0; mt < MT; ++mt) {
        const int arow = (mt * 16 + r) * DIM;
        a[mt].p[0] = *(const v8h*)&aTile[arow + k0 + h * 8];
        a[mt].p[1] = *(const v8h*)&aTile[arow + k0 + 16 + h * 8];
      }
      #pragma unroll
      for (int nt = 0; nt < NT; ++nt) {
        const size_t boff = (size_t)nt * 16 * DIM + k0 + h * 8;
        b[nt].p[0] = *(const v8h*)&bcol[boff];
        b[nt].p[1] = *(const v8h*)&bcol[boff + 16];
      }
      #pragma unroll
      for (int mt = 0; mt < MT; ++mt)
        #pragma unroll
        for (int nt = 0; nt < NT; ++nt)
          c[mt][nt] = __builtin_amdgcn_wmma_f32_16x16x32_f16(
                          false, a[mt].v, false, b[nt].v,
                          (short)0, c[mt][nt], false, false);
    }

    #pragma unroll
    for (int mt = 0; mt < MT; ++mt) {
      const int rBase = rowBase + mt * 16;
      const int pBase = (rBase + NROWS_HALF) & (NROWS - 1);
      #pragma unroll
      for (int nt = 0; nt < NT; ++nt) {
        const int colBase = (g * NT + nt) * 16;
        const bool isDiag = (colBase == rBase);
        const bool isPos  = (colBase == pBase);
        #pragma unroll
        for (int v = 0; v < 8; ++v) {
          const float s    = c[mt][nt][v] * INV_T;      // sim / T
          const int   m    = v + 8 * h;                 // M within tile
          const bool  self = (r == m);                  // N == M
          if (isPos && self) posArr[mt * 16 + m] = s;   // positive pair
          float e = __expf(s);
          if (isDiag && self) e = 0.0f;                 // mask diagonal
          partial[mt][v] += e;
        }
      }
    }
  }

  // Reduce across the 16 lanes holding each row (butterfly stays in-half).
  #pragma unroll
  for (int mt = 0; mt < MT; ++mt)
    #pragma unroll
    for (int v = 0; v < 8; ++v) {
      #pragma unroll
      for (int off = 1; off < 16; off <<= 1)
        partial[mt][v] += __shfl_xor(partial[mt][v], off, 32);
    }
  if (r == 0) {
    #pragma unroll
    for (int mt = 0; mt < MT; ++mt)
      #pragma unroll
      for (int v = 0; v < 8; ++v)
        waveSums[wave][mt * 16 + v + 8 * h] = partial[mt][v];
  }
  __syncthreads();

  // Fixed-order combine across the 8 waves -> lse - pos per row.
  if (tid < ROWS_PB) {
    float s = 0.0f;
    #pragma unroll
    for (int w = 0; w < 8; ++w) s += waveSums[w][tid];
    lmp[tid] = __logf(s) - posArr[tid];
  }
  __syncthreads();
  if (tid == 0) {
    float s = 0.0f;
    #pragma unroll
    for (int m = 0; m < ROWS_PB; ++m) s += lmp[m];
    blockLoss[rb] = s;
  }
}

// ---------------------------------------------------------------------------
// Kernel 3: deterministic final reduction of block losses -> mean loss.
// ---------------------------------------------------------------------------
__global__ void ntx_reduce(const float* __restrict__ blockLoss,
                           float* __restrict__ out) {
  const int lane = threadIdx.x;   // 32 threads
  float s = 0.0f;
  for (int i = lane; i < NRB; i += 32) s += blockLoss[i];
  #pragma unroll
  for (int off = 16; off > 0; off >>= 1)
    s += __shfl_xor(s, off, 32);
  if (lane == 0) out[0] = s * (1.0f / (float)NROWS);
}

extern "C" void kernel_launch(void* const* d_in, const int* in_sizes, int n_in,
                              void* d_out, int out_size, void* d_ws, size_t ws_size,
                              hipStream_t stream) {
  (void)in_sizes; (void)n_in; (void)out_size; (void)ws_size;

  const float* logits1 = (const float*)d_in[0];
  const float* logits2 = (const float*)d_in[1];
  float* out = (float*)d_out;

  // Workspace layout: [ z_f16 : 8192*1024*2 B ][ blockLoss : NRB f32 ]
  _Float16* z = (_Float16*)d_ws;
  float* blockLoss = (float*)((char*)d_ws + (size_t)NROWS * DIM * sizeof(_Float16));

  ntx_normalize<<<NROWS, 256, 0, stream>>>(logits1, logits2, z);
  ntx_gram_lse<<<NRB, 256, (size_t)ROWS_PB * DIM * sizeof(_Float16), stream>>>(z, blockLoss);
  ntx_reduce<<<1, 32, 0, stream>>>(blockLoss, out);
}
